// GAE_14001593385007
// MI455X (gfx1250) — matrix operand order repaired and checked by reading, not compiled, not run on hardware
//
#include <hip/hip_runtime.h>
#include <stdint.h>

// ---------------------------------------------------------------------------
// Types for CDNA5 WMMA
// ---------------------------------------------------------------------------
typedef __attribute__((ext_vector_type(16))) __bf16 v16bf;
typedef __attribute__((ext_vector_type(8)))  float  v8f;

__device__ __forceinline__ uint32_t f2bf(float f) {
  // round-to-nearest-even fp32 -> bf16 (bit trick)
  uint32_t u = __builtin_bit_cast(uint32_t, f);
  uint32_t r = u + 0x7FFFu + ((u >> 16) & 1u);
  return r >> 16;
}

__device__ __forceinline__ uint32_t pack2bf(float lo, float hi) {
  return f2bf(lo) | (f2bf(hi) << 16);
}

// ---------------------------------------------------------------------------
// Degree / normalization
// ---------------------------------------------------------------------------
__global__ void k_fill_deg(float* deg, int n) {
  int i = blockIdx.x * blockDim.x + threadIdx.x;
  if (i < n) deg[i] = 1.0f;   // self-loop contribution
}

__global__ void k_deg_accum(const int* __restrict__ dst, float* deg, int e) {
  int i = blockIdx.x * blockDim.x + threadIdx.x;
  if (i < e) atomicAdd(&deg[dst[i]], 1.0f);
}

__global__ void k_dinv(const float* __restrict__ deg, float* dinv, int n) {
  int i = blockIdx.x * blockDim.x + threadIdx.x;
  if (i < n) {
    float d = deg[i];
    dinv[i] = (d > 0.0f) ? rsqrtf(d) : 0.0f;
  }
}

__global__ void k_norm(const int* __restrict__ src, const int* __restrict__ dst,
                       const float* __restrict__ dinv, float* norm, int e) {
  int i = blockIdx.x * blockDim.x + threadIdx.x;
  if (i < e) norm[i] = dinv[src[i]] * dinv[dst[i]];
}

// ---------------------------------------------------------------------------
// Bulk fp32 -> bf16 conversion (row-major, element order preserved).
// One thread handles 4 floats -> 2 packed dwords.
// ---------------------------------------------------------------------------
__global__ void k_f32_to_bf16(const float* __restrict__ in,
                              uint32_t* __restrict__ out, long long n4) {
  long long t = (long long)blockIdx.x * blockDim.x + threadIdx.x;
  if (t >= n4) return;
  float4 f = *(const float4*)(in + t * 4);
  uint2 o;
  o.x = pack2bf(f.x, f.y);
  o.y = pack2bf(f.z, f.w);
  *(uint2*)(out + t * 2) = o;
}

// ---------------------------------------------------------------------------
// Pack fp32 weight matrix [K x N] into CDNA5 bf16 B-matrix fragments.
// Layout: dword index = (((kt*NT + nt)*32) + lane)*8 + v
//   lane 0-15 : N = nt*16 + lane,    K = kt*32 + 16*(v/4) + 2*(v%4), +1
//   lane 16-31: N = nt*16 + lane-16, K = kt*32 + 8 + 16*(v/4) + 2*(v%4), +1
// ---------------------------------------------------------------------------
__global__ void k_pack_w(const float* __restrict__ W, uint32_t* __restrict__ pack,
                         int KDIM, int NDIM) {
  int NT = NDIM >> 4;
  int total = (KDIM >> 5) * NT * 32 * 8;
  int i = blockIdx.x * blockDim.x + threadIdx.x;
  if (i >= total) return;
  int v    = i & 7;
  int lane = (i >> 3) & 31;
  int tile = i >> 8;            // kt*NT + nt
  int nt = tile % NT;
  int kt = tile / NT;
  int k = kt * 32 + ((lane >> 4) << 3) + ((v >> 2) << 4) + 2 * (v & 3);
  int c = nt * 16 + (lane & 15);
  pack[i] = pack2bf(W[(size_t)k * NDIM + c], W[(size_t)(k + 1) * NDIM + c]);
}

// ---------------------------------------------------------------------------
// WMMA GEMM: out[M x NDIM] = A[M x KDIM] @ Bpack, fp32 accumulate.
// A is bf16 row-major [M x KDIM]. One wave = one 16-row M tile x full NDIM.
// 256 threads = 8 M tiles per block. Inner loop is pure b128 loads + WMMA.
// ---------------------------------------------------------------------------
template <int KDIM, int NDIM>
__global__ __launch_bounds__(256) void k_gemm_wmma(
    const uint16_t* __restrict__ Abf, const uint32_t* __restrict__ Bpack,
    float* __restrict__ out, int nrows) {
  constexpr int KT = KDIM / 32;
  constexpr int NT = NDIM / 16;
  const int lane = threadIdx.x & 31;
  const int wave = threadIdx.x >> 5;
  const int m0 = (blockIdx.x * 8 + wave) * 16;
  if (m0 >= nrows) return;                 // wave-uniform exit: EXEC stays full
  const int half = lane >> 4;              // 0 or 1 (K half select)
  const int ln = lane & 15;
  const int rA = min(m0 + ln, nrows - 1);  // clamp loads, keep EXEC all-1s

  v8f acc[NT] = {};

  union BF { uint32_t u[8]; v16bf v; };

#pragma unroll
  for (int kt = 0; kt < KT; ++kt) {
    const int k0 = kt * 32 + half * 8;
    // A fragment (16-bit A 16x32 layout): lane half selects K runs
    //   {k0..k0+7} and {k0+16..k0+23}, 8 bf16 each -> two b128 loads.
    const uint32_t* ap = (const uint32_t*)(Abf + (size_t)rA * KDIM + k0);
    BF afr;
    uint4 q0 = *(const uint4*)(ap + 0);
    uint4 q1 = *(const uint4*)(ap + 8);
    afr.u[0] = q0.x; afr.u[1] = q0.y; afr.u[2] = q0.z; afr.u[3] = q0.w;
    afr.u[4] = q1.x; afr.u[5] = q1.y; afr.u[6] = q1.z; afr.u[7] = q1.w;

    const uint32_t* bbase = Bpack + ((size_t)kt * NT * 32 + lane) * 8;
#pragma unroll
    for (int nt = 0; nt < NT; ++nt) {
      BF bfr;
      const uint32_t* b = bbase + (size_t)nt * 256;
      uint4 r0 = *(const uint4*)(b + 0);
      uint4 r1 = *(const uint4*)(b + 4);
      bfr.u[0] = r0.x; bfr.u[1] = r0.y; bfr.u[2] = r0.z; bfr.u[3] = r0.w;
      bfr.u[4] = r1.x; bfr.u[5] = r1.y; bfr.u[6] = r1.z; bfr.u[7] = r1.w;
      acc[nt] = __builtin_amdgcn_wmma_f32_16x16x32_bf16(
          false, afr.v, false, bfr.v, (short)0, acc[nt], false, false);
    }
  }

  // C/D layout: lane holds col N = ln, rows M = half*8 + v
  const int rbase = m0 + half * 8;
#pragma unroll
  for (int nt = 0; nt < NT; ++nt) {
#pragma unroll
    for (int v = 0; v < 8; ++v) {
      int r = rbase + v;
      if (r < nrows) out[(size_t)r * NDIM + nt * 16 + ln] = acc[nt][v];
    }
  }
}

// ---------------------------------------------------------------------------
// out[i,c] = feat[i,c] * dinv[i]^2 (+ bias[c])   (self-loop term + init)
// ---------------------------------------------------------------------------
template <int F>
__global__ void k_self_init(const float* __restrict__ feat,
                            const float* __restrict__ dinv,
                            const float* __restrict__ bias,
                            float* __restrict__ out, int nnodes) {
  long long t = (long long)blockIdx.x * blockDim.x + threadIdx.x;
  if (t >= (long long)nnodes * F) return;
  int i = (int)(t / F);
  int c = (int)(t % F);
  float di = dinv[i];
  float v = feat[t] * di * di;
  if (bias) v += bias[c];
  out[t] = v;
}

// ---------------------------------------------------------------------------
// Edge scatter-add: out[dst] += feat[src] * norm[e]   (float4 per thread)
// ---------------------------------------------------------------------------
template <int F>
__global__ __launch_bounds__(256) void k_edge_agg(
    const float* __restrict__ feat, const int* __restrict__ src,
    const int* __restrict__ dst, const float* __restrict__ norm,
    float* out, int nedges) {
  constexpr int C4 = F / 4;
  long long t = (long long)blockIdx.x * blockDim.x + threadIdx.x;
  long long e = t / C4;
  if (e >= nedges) return;
  int c = (int)(t % C4) * 4;
  int s = src[e], d = dst[e];
  float w = norm[e];
  float4 m = *(const float4*)(feat + (size_t)s * F + c);
  float* o = out + (size_t)d * F + c;
  atomicAdd(o + 0, m.x * w);
  atomicAdd(o + 1, m.y * w);
  atomicAdd(o + 2, m.z * w);
  atomicAdd(o + 3, m.w * w);
}

// ---------------------------------------------------------------------------
// h2 = bf16(relu(agg + b1)), two elements per thread -> packed dword store
// ---------------------------------------------------------------------------
__global__ void k_relu_bias_bf16(const float* __restrict__ agg,
                                 const float* __restrict__ b1,
                                 uint32_t* __restrict__ h2, long long n2) {
  long long t = (long long)blockIdx.x * blockDim.x + threadIdx.x;
  if (t >= n2) return;
  int c = (int)((t * 2) & 127);
  float2 a = *(const float2*)(agg + t * 2);
  float lo = fmaxf(a.x + b1[c], 0.0f);
  float hi = fmaxf(a.y + b1[c + 1], 0.0f);
  h2[t] = pack2bf(lo, hi);
}

// ---------------------------------------------------------------------------
// Host side
// ---------------------------------------------------------------------------
static inline size_t align_up(size_t x, size_t a) { return (x + a - 1) & ~(a - 1); }
static inline int cdiv(long long a, long long b) { return (int)((a + b - 1) / b); }

extern "C" void kernel_launch(void* const* d_in, const int* in_sizes, int n_in,
                              void* d_out, int out_size, void* d_ws, size_t ws_size,
                              hipStream_t stream) {
  constexpr int IN = 256, HID = 128, OUT = 64;
  const float* x  = (const float*)d_in[0];
  const int*   ei = (const int*)d_in[1];
  const float* W1 = (const float*)d_in[2];
  const float* b1 = (const float*)d_in[3];
  const float* W2 = (const float*)d_in[4];
  const float* b2 = (const float*)d_in[5];
  float* z = (float*)d_out;

  const int N = in_sizes[0] / IN;
  const int E = in_sizes[1] / 2;
  const int* src = ei;
  const int* dst = ei + E;

  // workspace carve-up
  char* ws = (char*)d_ws;
  size_t off = 0;
  auto carve = [&](size_t bytes) {
    char* p = ws + off;
    off = align_up(off + bytes, 256);
    return p;
  };
  float*    deg  = (float*)carve((size_t)N * 4);
  float*    dinv = (float*)carve((size_t)N * 4);
  float*    nrm  = (float*)carve((size_t)E * 4);
  uint32_t* w1p  = (uint32_t*)carve((size_t)(IN / 32) * (HID / 16) * 256 * 4);
  uint32_t* w2p  = (uint32_t*)carve((size_t)(HID / 32) * (OUT / 16) * 256 * 4);
  uint16_t* xb   = (uint16_t*)carve((size_t)N * IN * 2);  // x in bf16
  float*    bufA = (float*)carve((size_t)N * HID * 4);    // h1 fp32; reused: h2 bf16
  float*    bufB = (float*)carve((size_t)N * HID * 4);    // agg1 fp32; reused: t2 fp32
  (void)ws_size; (void)n_in; (void)out_size;

  const int TPB = 256;
  const int mblocks = cdiv(cdiv(N, 16), 8);

  // 1) symmetric normalization
  k_fill_deg<<<cdiv(N, TPB), TPB, 0, stream>>>(deg, N);
  k_deg_accum<<<cdiv(E, TPB), TPB, 0, stream>>>(dst, deg, E);
  k_dinv<<<cdiv(N, TPB), TPB, 0, stream>>>(deg, dinv, N);
  k_norm<<<cdiv(E, TPB), TPB, 0, stream>>>(src, dst, dinv, nrm, E);

  // 2) pack weights into WMMA B-fragment layout; pre-convert x to bf16
  k_pack_w<<<cdiv((IN / 32) * (HID / 16) * 256, TPB), TPB, 0, stream>>>(W1, w1p, IN, HID);
  k_pack_w<<<cdiv((HID / 32) * (OUT / 16) * 256, TPB), TPB, 0, stream>>>(W2, w2p, HID, OUT);
  k_f32_to_bf16<<<cdiv((long long)N * IN / 4, TPB), TPB, 0, stream>>>(
      x, (uint32_t*)xb, (long long)N * IN / 4);

  // 3) layer 1: h1 = x @ W1 (WMMA bf16 -> fp32)
  k_gemm_wmma<IN, HID><<<mblocks, TPB, 0, stream>>>(xb, w1p, bufA, N);

  // 4) aggregation 1: agg1 = selfloop + scatter-add
  k_self_init<HID><<<cdiv((long long)N * HID, TPB), TPB, 0, stream>>>(bufA, dinv, nullptr, bufB, N);
  k_edge_agg<HID><<<cdiv((long long)E * (HID / 4), TPB), TPB, 0, stream>>>(bufA, src, dst, nrm, bufB, E);

  // 5) h2 = bf16(relu(agg1 + b1))  (stored over bufA)
  uint16_t* h2 = (uint16_t*)bufA;
  k_relu_bias_bf16<<<cdiv((long long)N * HID / 2, TPB), TPB, 0, stream>>>(
      bufB, b1, (uint32_t*)h2, (long long)N * HID / 2);

  // 6) layer 2: t2 = h2 @ W2 (WMMA bf16 -> fp32), overwrites bufB
  k_gemm_wmma<HID, OUT><<<mblocks, TPB, 0, stream>>>(h2, w2p, bufB, N);

  // 7) aggregation 2 straight into d_out: z = b2 + selfloop + scatter-add
  k_self_init<OUT><<<cdiv((long long)N * OUT, TPB), TPB, 0, stream>>>(bufB, dinv, b2, z, N);
  k_edge_agg<OUT><<<cdiv((long long)E * (OUT / 4), TPB), TPB, 0, stream>>>(bufB, src, dst, nrm, z, E);
}
